// PWSAComplete_24567212933764
// MI455X (gfx1250) — compile-verified, matrix operand order checked by reading, not run on hardware
//
#include <hip/hip_runtime.h>
#include <math.h>

typedef __attribute__((ext_vector_type(16))) _Float16 v16h;
typedef __attribute__((ext_vector_type(8)))  float    v8f;
typedef __attribute__((ext_vector_type(4)))  float    v4f;

static constexpr int kB = 32, kC = 7, kBC = 224, kSeq = 1024, kPred = 192;
static constexpr int kDm = 256, kNh = 8, kDhd = 32;     // wrat attention
static constexpr int kPL = 16, kP = 128, kL = 64;
static constexpr int kRows = kBC * kL;                  // 14336
static constexpr int kNst = 16;                         // D_STATE
static constexpr int kCut = 256;

// ---------------------------------------------------------------------------
// WMMA GEMM: Y[M,N] = X[M,K] @ W[N,K]^T (+bias).  f32 in/out, f16 multiply,
// f32 accumulate via v_wmma_f32_16x16x32_f16.  Each wave computes a 16x64
// tile (4 independent WMMAs / k-step, A-fragment reused 4x) so the f16
// conversions for the next fragments co-execute with the matrix pipe.
// Requires M%16==0, K%32==0; any N (W row index clamped for partial tiles,
// invalid columns discarded at store so EXEC is all-ones during WMMA).
// ---------------------------------------------------------------------------
__device__ __forceinline__ v16h frag_load_f16(const float* __restrict__ p) {
  // ISA 7.12.2 16-bit A/B layout: this lane holds k = kb + {0..7} (halves 0-7)
  // and k = kb + 16 + {0..7} (halves 8-15); caller pre-offsets p by kb.
  const v4f* q = (const v4f*)p;
  v4f a0 = q[0], a1 = q[1], a2 = q[4], a3 = q[5];
  v16h f;
  f[0] = (_Float16)a0[0]; f[1] = (_Float16)a0[1]; f[2] = (_Float16)a0[2]; f[3] = (_Float16)a0[3];
  f[4] = (_Float16)a1[0]; f[5] = (_Float16)a1[1]; f[6] = (_Float16)a1[2]; f[7] = (_Float16)a1[3];
  f[8] = (_Float16)a2[0]; f[9] = (_Float16)a2[1]; f[10] = (_Float16)a2[2]; f[11] = (_Float16)a2[3];
  f[12] = (_Float16)a3[0]; f[13] = (_Float16)a3[1]; f[14] = (_Float16)a3[2]; f[15] = (_Float16)a3[3];
  return f;
}

__global__ void k_gemm(const float* __restrict__ X, const float* __restrict__ W,
                       const float* __restrict__ bias, float* __restrict__ Y,
                       int M, int N, int K, int ldx, int ldw, int ldy) {
  int wave = threadIdx.x >> 5;
  int lane = threadIdx.x & 31;
  int nt0 = (blockIdx.y * 4 + wave) * 4;     // first 16-col subtile of this wave
  if (nt0 * 16 >= N) return;                 // wave-uniform guard: EXEC stays full
  int mt = blockIdx.x;
  int kb = (lane & 16) ? 8 : 0;
  int l15 = lane & 15;
  const float* xp = X + (size_t)(mt * 16 + l15) * ldx + kb;
  const float* wp[4];
#pragma unroll
  for (int s = 0; s < 4; ++s) {
    int wr = nt0 * 16 + s * 16 + l15;
    if (wr > N - 1) wr = N - 1;              // clamp: garbage acc, discarded at store
    wp[s] = W + (size_t)wr * ldw + kb;
  }
  v8f acc[4] = {};
  for (int k0 = 0; k0 < K; k0 += 32) {
    __builtin_prefetch(xp + k0 + 128, 0, 1);     // global_prefetch_b8: next X tiles
    __builtin_prefetch(wp[0] + k0 + 128, 0, 1);  // next W tiles
    v16h a  = frag_load_f16(xp + k0);
    v16h b0 = frag_load_f16(wp[0] + k0);
    v16h b1 = frag_load_f16(wp[1] + k0);
    v16h b2 = frag_load_f16(wp[2] + k0);
    v16h b3 = frag_load_f16(wp[3] + k0);
    acc[0] = __builtin_amdgcn_wmma_f32_16x16x32_f16(false, a, false, b0, (short)0, acc[0], false, false);
    acc[1] = __builtin_amdgcn_wmma_f32_16x16x32_f16(false, a, false, b1, (short)0, acc[1], false, false);
    acc[2] = __builtin_amdgcn_wmma_f32_16x16x32_f16(false, a, false, b2, (short)0, acc[2], false, false);
    acc[3] = __builtin_amdgcn_wmma_f32_16x16x32_f16(false, a, false, b3, (short)0, acc[3], false, false);
  }
  int m0 = mt * 16 + ((lane & 16) ? 8 : 0);  // D layout: lanes>=16 hold rows 8..15
#pragma unroll
  for (int s = 0; s < 4; ++s) {
    int n = nt0 * 16 + s * 16 + l15;
    if (n < N) {
      float bv = bias ? bias[n] : 0.0f;
#pragma unroll
      for (int r = 0; r < 8; ++r)
        Y[(size_t)(m0 + r) * ldy + n] = acc[s][r] + bv;
    }
  }
}

// ---------------------------------------------------------------------------
// block reduction helper (power-of-two blockDim)
// ---------------------------------------------------------------------------
__device__ __forceinline__ float block_sum(float v, float* sb) {
  int tid = threadIdx.x;
  sb[tid] = v;
  __syncthreads();
  for (int s = blockDim.x >> 1; s > 0; s >>= 1) {
    if (tid < s) sb[tid] += sb[tid + s];
    __syncthreads();
  }
  float r = sb[0];
  __syncthreads();
  return r;
}

__device__ __forceinline__ float sigm(float v) { return 1.0f / (1.0f + expf(-v)); }

// ---------------------------------------------------------------------------
// RevIN: per (b,c) mean/std over SEQ, normalize + affine
// ---------------------------------------------------------------------------
__global__ void k_revin(const float* __restrict__ x, const float* __restrict__ rw,
                        const float* __restrict__ rb, float* __restrict__ xn,
                        float* __restrict__ meanA, float* __restrict__ stdA) {
  __shared__ float sb[256];
  int bc = blockIdx.x, tid = threadIdx.x, c = bc % kC;
  float v[4], loc = 0.f;
#pragma unroll
  for (int r = 0; r < 4; ++r) { v[r] = x[(size_t)bc * kSeq + tid + r * 256]; loc += v[r]; }
  float mean = block_sum(loc, sb) * (1.0f / kSeq);
  loc = 0.f;
#pragma unroll
  for (int r = 0; r < 4; ++r) { float d = v[r] - mean; loc += d * d; }
  float sd = sqrtf(block_sum(loc, sb) * (1.0f / kSeq) + 1e-5f);
  if (tid == 0) { meanA[bc] = mean; stdA[bc] = sd; }
  float w = rw[c], b = rb[c];
#pragma unroll
  for (int r = 0; r < 4; ++r)
    xn[(size_t)bc * kSeq + tid + r * 256] = (v[r] - mean) / sd * w + b;
}

// ---------------------------------------------------------------------------
// Variable attention core over C=7 tokens (4 heads, dh=16)
// ---------------------------------------------------------------------------
__global__ void k_va_attn(const float* __restrict__ qkv, float* __restrict__ ctx) {
  int idx = blockIdx.x * blockDim.x + threadIdx.x;
  if (idx >= kB * kC * 4) return;
  int h = idx & 3, c = (idx >> 2) % kC, b = idx / (kC * 4);
  const float* qp = qkv + (size_t)(b * kC + c) * 192 + h * 16;
  float q[16];
#pragma unroll
  for (int d = 0; d < 16; ++d) q[d] = qp[d];
  float s[7], mx = -1e30f;
  for (int j = 0; j < kC; ++j) {
    const float* kp = qkv + (size_t)(b * kC + j) * 192 + 64 + h * 16;
    float t = 0.f;
#pragma unroll
    for (int d = 0; d < 16; ++d) t += q[d] * kp[d];
    s[j] = t * 0.25f;                          // 1/sqrt(16)
    mx = fmaxf(mx, s[j]);
  }
  float sum = 0.f;
  for (int j = 0; j < kC; ++j) { s[j] = expf(s[j] - mx); sum += s[j]; }
  float o[16];
#pragma unroll
  for (int d = 0; d < 16; ++d) o[d] = 0.f;
  for (int j = 0; j < kC; ++j) {
    float wgt = s[j] / sum;
    const float* vp = qkv + (size_t)(b * kC + j) * 192 + 128 + h * 16;
#pragma unroll
    for (int d = 0; d < 16; ++d) o[d] += wgt * vp[d];
  }
  float* op = ctx + (size_t)(b * kC + c) * 64 + h * 16;
#pragma unroll
  for (int d = 0; d < 16; ++d) op[d] = o[d];
}

// xv = LN_1024(xn + sigmoid(gate)*outY) * ln_w + ln_b
__global__ void k_va_fin(const float* __restrict__ xn, const float* __restrict__ outY,
                         const float* __restrict__ gate1, const float* __restrict__ lnw,
                         const float* __restrict__ lnb, float* __restrict__ xv) {
  __shared__ float sb[256];
  int bc = blockIdx.x, tid = threadIdx.x;
  float g = sigm(gate1[0]);
  float t[4], loc = 0.f;
#pragma unroll
  for (int r = 0; r < 4; ++r) {
    int l = tid + r * 256;
    t[r] = xn[(size_t)bc * kSeq + l] + g * outY[(size_t)bc * kSeq + l];
    loc += t[r];
  }
  float mean = block_sum(loc, sb) * (1.0f / kSeq);
  loc = 0.f;
#pragma unroll
  for (int r = 0; r < 4; ++r) { float d = t[r] - mean; loc += d * d; }
  float inv = rsqrtf(block_sum(loc, sb) * (1.0f / kSeq) + 1e-5f);
#pragma unroll
  for (int r = 0; r < 4; ++r) {
    int l = tid + r * 256;
    xv[(size_t)bc * kSeq + l] = (t[r] - mean) * inv * lnw[l] + lnb[l];
  }
}

__global__ void k_mixsm(const float* __restrict__ A, float* __restrict__ S) {
  int r = threadIdx.x;
  if (r >= kC) return;
  float mx = -1e30f;
  for (int j = 0; j < kC; ++j) mx = fmaxf(mx, A[r * kC + j]);
  float e[7], sum = 0.f;
  for (int j = 0; j < kC; ++j) { e[j] = expf(A[r * kC + j] - mx); sum += e[j]; }
  for (int j = 0; j < kC; ++j) S[r * kC + j] = e[j] / sum;
}

__global__ void k_mix(const float* __restrict__ xv, const float* __restrict__ smA,
                      float* __restrict__ xm) {
  int b = blockIdx.x, i = blockIdx.y;
  float a[7];
#pragma unroll
  for (int j = 0; j < kC; ++j) a[j] = smA[i * kC + j];
  for (int l = threadIdx.x; l < kSeq; l += 256) {
    float s = 0.f;
#pragma unroll
    for (int j = 0; j < kC; ++j) s += a[j] * xv[(size_t)(b * kC + j) * kSeq + l];
    xm[(size_t)(b * kC + i) * kSeq + l] = s;
  }
}

// patch embed: emb[bc,p,d] = sum_k up[p*8+k] * pw[d,k] + pb[d]   (K=16 -> plain)
__global__ void k_patch(const float* __restrict__ xm, const float* __restrict__ pw,
                        const float* __restrict__ pb, float* __restrict__ emb) {
  __shared__ float su[16];
  int blk = blockIdx.x, p = blk & (kP - 1), bc = blk >> 7, d = threadIdx.x;
  if (d < 16) {
    int idx = p * 8 + d;
    if (idx > kSeq - 1) idx = kSeq - 1;       // up = concat(u, repeat(last,8))
    su[d] = xm[(size_t)bc * kSeq + idx];
  }
  __syncthreads();
  float acc = pb[d];
#pragma unroll
  for (int k = 0; k < 16; ++k) acc += su[k] * pw[d * 16 + k];
  emb[((size_t)(bc * kP + p)) * kDm + d] = acc;
}

// DWT: LL/LH[bc,t,d] = sum_k emb[bc, 2t+k-1, d] * w[d,k]   (stride2, pad1, FL=4)
__global__ void k_dwt(const float* __restrict__ emb, const float* __restrict__ hw,
                      const float* __restrict__ gw, float* __restrict__ LL,
                      float* __restrict__ LH) {
  int blk = blockIdx.x, t = blk & (kL - 1), bc = blk >> 6, d = threadIdx.x;
  float a = 0.f, b = 0.f;
#pragma unroll
  for (int k = 0; k < 4; ++k) {
    int j = 2 * t + k - 1;
    if (j >= 0 && j < kP) {
      float e = emb[((size_t)(bc * kP + j)) * kDm + d];
      a += e * hw[d * 4 + k];
      b += e * gw[d * 4 + k];
    }
  }
  LL[((size_t)(bc * kL + t)) * kDm + d] = a;
  LH[((size_t)(bc * kL + t)) * kDm + d] = b;
}

// He = gelu(conv1(|LH|, env_w, pad1) + env_b), exact erf gelu
__global__ void k_env(const float* __restrict__ LH, const float* __restrict__ ew,
                      const float* __restrict__ eb, float* __restrict__ He) {
  int blk = blockIdx.x, t = blk & (kL - 1), bc = blk >> 6, d = threadIdx.x;
  float a = eb[d];
#pragma unroll
  for (int k = 0; k < 3; ++k) {
    int j = t + k - 1;
    if (j >= 0 && j < kL) a += fabsf(LH[((size_t)(bc * kL + j)) * kDm + d]) * ew[d * 3 + k];
  }
  He[((size_t)(bc * kL + t)) * kDm + d] = 0.5f * a * (1.0f + erff(a * 0.70710678118654752f));
}

// gate[row] = sigmoid((mean_d |He| - tau) * 10)
__global__ void k_gate(const float* __restrict__ He, const float* __restrict__ rawTau,
                       float* __restrict__ gate) {
  __shared__ float sb[256];
  int row = blockIdx.x, d = threadIdx.x;
  float m = block_sum(fabsf(He[(size_t)row * kDm + d]), sb) * (1.0f / kDm);
  if (d == 0) {
    float tau = sigm(rawTau[0]);
    gate[row] = sigm((m - tau) * 10.0f);
  }
}

// full attention core: per (bc, head), 64 queries x 64 keys x 32 dims.
// gate (nullable) multiplies softmax weights per KEY (reference semantics).
__global__ void k_attn(const float* __restrict__ Q, const float* __restrict__ K,
                       const float* __restrict__ V, const float* __restrict__ gate,
                       float* __restrict__ ctx) {
  __shared__ float sc[64 * 65];
  int blk = blockIdx.x, bc = blk >> 3, h = blk & 7, i = threadIdx.x;
  const float* qp = Q + ((size_t)(bc * kL + i)) * kDm + h * kDhd;
  float q[32];
#pragma unroll
  for (int d = 0; d < 32; ++d) q[d] = qp[d];
  const float scale = 0.17677669529663687f;  // 1/sqrt(32)
  for (int j = 0; j < 64; ++j) {
    const float* kp = K + ((size_t)(bc * kL + j)) * kDm + h * kDhd;
    float s = 0.f;
#pragma unroll
    for (int d = 0; d < 32; ++d) s += q[d] * kp[d];
    sc[i * 65 + j] = s * scale;
  }
  float mx = -1e30f;
  for (int j = 0; j < 64; ++j) mx = fmaxf(mx, sc[i * 65 + j]);
  float sum = 0.f;
  for (int j = 0; j < 64; ++j) { float e = expf(sc[i * 65 + j] - mx); sc[i * 65 + j] = e; sum += e; }
  float inv = 1.0f / sum;
  if (gate) {
    for (int j = 0; j < 64; ++j) sc[i * 65 + j] *= inv * gate[bc * kL + j];
  } else {
    for (int j = 0; j < 64; ++j) sc[i * 65 + j] *= inv;
  }
  float o[32];
#pragma unroll
  for (int d = 0; d < 32; ++d) o[d] = 0.f;
  for (int j = 0; j < 64; ++j) {
    float wgt = sc[i * 65 + j];
    const float* vp = V + ((size_t)(bc * kL + j)) * kDm + h * kDhd;
#pragma unroll
    for (int d = 0; d < 32; ++d) o[d] += wgt * vp[d];
  }
  float* op = ctx + ((size_t)(bc * kL + i)) * kDm + h * kDhd;
#pragma unroll
  for (int d = 0; d < 32; ++d) op[d] = o[d];
}

// fused = tpY * sigmoid(gpY) + Ls
__global__ void k_fused(const float* __restrict__ tp, const float* __restrict__ gp,
                        const float* __restrict__ Ls, float* __restrict__ out) {
  size_t i = (size_t)blockIdx.x * 256 + threadIdx.x;
  out[i] = tp[i] * sigm(gp[i]) + Ls[i];
}

// generic LN over D=256: out = LN(A (+Bb)) * w + b (+ addend)
__global__ void k_ln(const float* __restrict__ A, const float* __restrict__ Bb,
                     const float* __restrict__ w, const float* __restrict__ b,
                     const float* __restrict__ addend, float* __restrict__ out) {
  __shared__ float sb[256];
  int row = blockIdx.x, d = threadIdx.x;
  size_t idx = (size_t)row * kDm + d;
  float v = A[idx] + (Bb ? Bb[idx] : 0.0f);
  float m = block_sum(v, sb) * (1.0f / kDm);
  float dv = v - m;
  float var = block_sum(dv * dv, sb) * (1.0f / kDm);
  float o = dv * rsqrtf(var + 1e-5f) * w[d] + b[d];
  out[idx] = addend ? (o + addend[idx]) : o;
}

__global__ void k_softplus(float* __restrict__ v) {
  size_t i = (size_t)blockIdx.x * 256 + threadIdx.x;
  float x = v[i];
  v[i] = (x > 20.0f) ? x : log1pf(expf(x));
}

// selective-scan: one thread per (bc, d); 16-state recurrence over 64 steps.
// xz: [rows,512] (x | z), proj: [rows,288] (dr | Bs | Cs), delta: [rows,256]
__global__ void k_scan(const float* __restrict__ delta, const float* __restrict__ xz,
                       const float* __restrict__ proj, const float* __restrict__ aLog,
                       const float* __restrict__ Dp, float* __restrict__ ssmT) {
  int bc = blockIdx.x, d = threadIdx.x;
  float A[kNst], h[kNst];
#pragma unroll
  for (int n = 0; n < kNst; ++n) { A[n] = -expf(aLog[d * kNst + n]); h[n] = 0.f; }
  float Dd = Dp[d];
  for (int t = 0; t < kL; ++t) {
    size_t row = (size_t)bc * kL + t;
    float de = delta[row * kDm + d];
    float xvv = xz[row * 512 + d];
    float zv = xz[row * 512 + 256 + d];
    float dx = de * xvv;
    float y = 0.f;
#pragma unroll
    for (int n = 0; n < kNst; ++n) {
      h[n] = expf(de * A[n]) * h[n] + dx * proj[row * 288 + 256 + n];
      y += h[n] * proj[row * 288 + 272 + n];
    }
    y += xvv * Dd;
    ssmT[row * kDm + d] = y * (zv * sigm(zv));   // y * silu(z)
  }
}

// inverse DWT (transposed depthwise conv, lhs_dilation=2, pad FL-1-PV=2)
// rec[bc, d*128+p] = sum_k h[d,3-k]*Lf[t,d] + g[d,3-k]*Hf[t,d], j=p+k-2=2t
__global__ void k_idwt(const float* __restrict__ Lf, const float* __restrict__ Hf,
                       const float* __restrict__ hw, const float* __restrict__ gw,
                       float* __restrict__ rec) {
  int blk = blockIdx.x, p = blk & (kP - 1), bc = blk >> 7, d = threadIdx.x;
  float acc = 0.f;
#pragma unroll
  for (int k = 0; k < 4; ++k) {
    int j = p + k - 2;
    if (j >= 0 && (j & 1) == 0) {
      int t = j >> 1;
      if (t < kL) {
        size_t ri = ((size_t)(bc * kL + t)) * kDm + d;
        acc += hw[d * 4 + (3 - k)] * Lf[ri] + gw[d * 4 + (3 - k)] * Hf[ri];
      }
    }
  }
  rec[(size_t)bc * (kDm * kP) + d * kP + p] = acc;
}

// direct rfft(1024)[:256]: feat = [Re | Im].  Exact mod-1024 phase reduction.
__global__ void k_dft(const float* __restrict__ xm, float* __restrict__ feat) {
  __shared__ float su[kSeq];
  int bc = blockIdx.x, k = threadIdx.x;
  for (int n = k; n < kSeq; n += 256) su[n] = xm[(size_t)bc * kSeq + n];
  __syncthreads();
  float re = 0.f, im = 0.f;
  for (int n = 0; n < kSeq; ++n) {
    int m = (k * n) & (kSeq - 1);
    float ang = -6.283185307179586f * (float)m * (1.0f / kSeq);
    float s, c;
    sincosf(ang, &s, &c);
    re += su[n] * c;
    im += su[n] * s;
  }
  feat[(size_t)bc * (2 * kCut) + k] = re;
  feat[(size_t)bc * (2 * kCut) + kCut + k] = im;
}

__global__ void k_final(const float* __restrict__ yhat, const float* __restrict__ fitsY,
                        const float* __restrict__ rw, const float* __restrict__ rb,
                        const float* __restrict__ meanA, const float* __restrict__ stdA,
                        float* __restrict__ out) {
  int bc = blockIdx.x, o = threadIdx.x, c = bc % kC;
  float y = yhat[(size_t)bc * kPred + o] + fitsY[(size_t)bc * kPred + o];
  y = (y - rb[c]) / (rw[c] + 1e-10f);
  out[(size_t)bc * kPred + o] = y * stdA[bc] + meanA[bc];
}

// ---------------------------------------------------------------------------
// input indices (setup_inputs() insertion order, depth-first)
// ---------------------------------------------------------------------------
enum {
  IN_X = 0, RV_W, RV_B,
  VA_INW, VA_INB, VA_QKVW, VA_QKVB, VA_AOW, VA_AOB, VA_OUTW, VA_OUTB,
  VA_LNW, VA_LNB, VA_GATE, MIXER_A, PATCH_W, PATCH_B, DWT_H, DWT_G,
  RAW_TAU, ENV_W, ENV_B,
  ALL_QW, ALL_QB, ALL_KW, ALL_KB, ALL_VW, ALL_VB, ALL_OW, ALL_OB,
  ALH_QW, ALH_QB, ALH_KW, ALH_KB, ALH_VW, ALH_VB, ALH_OW, ALH_OB,
  TP_W, TP_B, GP_W, GP_B,
  S1_INW, S1_XW, S1_DTW, S1_DTB, S1_ALOG, S1_D, S1_OUTW, S1_LNW, S1_LNB,
  S2_INW, S2_XW, S2_DTW, S2_DTB, S2_ALOG, S2_D, S2_OUTW, S2_LNW, S2_LNB,
  N1_W, N1_B, N2_W, N2_B, HEAD_W, HEAD_B, FITS_W, FITS_B
};

static inline void gemm(hipStream_t s, const float* X, const float* W, const float* b,
                        float* Y, int M, int N, int K, int ldx, int ldw, int ldy) {
  dim3 grid(M / 16, (N + 255) / 256);   // 4 waves/block, 16x64 tile per wave
  k_gemm<<<grid, 128, 0, s>>>(X, W, b, Y, M, N, K, ldx, ldw, ldy);
}

extern "C" void kernel_launch(void* const* d_in, const int* in_sizes, int n_in,
                              void* d_out, int out_size, void* d_ws, size_t ws_size,
                              hipStream_t stream) {
  (void)in_sizes; (void)n_in; (void)out_size; (void)ws_size;
  auto F = [&](int i) { return (const float*)d_in[i]; };

  // workspace bump allocator (64-float alignment for b128 vector loads)
  float* base = (float*)d_ws;
  size_t off = 0;
  auto A = [&](size_t n) { float* p = base + off; off += (n + 63) & ~(size_t)63; return p; };

  float* meanA = A(kBC);
  float* stdA  = A(kBC);
  float* xn    = A((size_t)kBC * kSeq);
  float* tok   = A((size_t)kBC * 64);
  float* qkvB  = A((size_t)kBC * 192);
  float* ctxVA = A((size_t)kBC * 64);
  float* aoY   = A((size_t)kBC * 64);
  float* outY  = A((size_t)kBC * kSeq);
  float* xv    = A((size_t)kBC * kSeq);
  float* smA   = A(64);
  float* xm    = A((size_t)kBC * kSeq);
  float* gateA = A(kRows);
  float* yhat  = A((size_t)kBC * kPred);
  float* feat  = A((size_t)kBC * 2 * kCut);
  float* fitsY = A((size_t)kBC * kPred);
  float* BIG0  = A((size_t)kRows * 512);     // emb (=BC*P*DM) / xz / recT
  float* P0    = A((size_t)kRows * 288);     // proj
  float* T0 = A((size_t)kRows * kDm); float* T1 = A((size_t)kRows * kDm);
  float* T2 = A((size_t)kRows * kDm); float* T3 = A((size_t)kRows * kDm);
  float* T4 = A((size_t)kRows * kDm); float* T5 = A((size_t)kRows * kDm);
  float* T6 = A((size_t)kRows * kDm); float* T7 = A((size_t)kRows * kDm);

  // ---- RevIN + variable attention ----
  k_revin<<<kBC, 256, 0, stream>>>(F(IN_X), F(RV_W), F(RV_B), xn, meanA, stdA);
  gemm(stream, xn, F(VA_INW), F(VA_INB), tok, kBC, 64, kSeq, kSeq, kSeq, 64);
  gemm(stream, tok, F(VA_QKVW), F(VA_QKVB), qkvB, kBC, 192, 64, 64, 64, 192);
  k_va_attn<<<(kB * kC * 4 + 127) / 128, 128, 0, stream>>>(qkvB, ctxVA);
  gemm(stream, ctxVA, F(VA_AOW), F(VA_AOB), aoY, kBC, 64, 64, 64, 64, 64);
  gemm(stream, aoY, F(VA_OUTW), F(VA_OUTB), outY, kBC, kSeq, 64, 64, 64, kSeq);
  k_va_fin<<<kBC, 256, 0, stream>>>(xn, outY, F(VA_GATE), F(VA_LNW), F(VA_LNB), xv);
  k_mixsm<<<1, 32, 0, stream>>>(F(MIXER_A), smA);
  k_mix<<<dim3(kB, kC), 256, 0, stream>>>(xv, smA, xm);

  // ---- patching, embed, DWT, envelope, gate ----
  float* emb = BIG0;
  k_patch<<<kBC * kP, kDm, 0, stream>>>(xm, F(PATCH_W), F(PATCH_B), emb);
  float* LL = T0; float* LH = T1;
  k_dwt<<<kBC * kL, kDm, 0, stream>>>(emb, F(DWT_H), F(DWT_G), LL, LH);
  float* He = T2;
  k_env<<<kBC * kL, kDm, 0, stream>>>(LH, F(ENV_W), F(ENV_B), He);
  k_gate<<<kRows, kDm, 0, stream>>>(He, F(RAW_TAU), gateA);

  // ---- aLL attention (on Ls = LL) ----
  float* Qb = T1; float* Kb = T3; float* Vb = T4; float* ctxA = T5; float* Lo = T6;
  gemm(stream, LL, F(ALL_QW), F(ALL_QB), Qb, kRows, kDm, kDm, kDm, kDm, kDm);
  gemm(stream, LL, F(ALL_KW), F(ALL_KB), Kb, kRows, kDm, kDm, kDm, kDm, kDm);
  gemm(stream, LL, F(ALL_VW), F(ALL_VB), Vb, kRows, kDm, kDm, kDm, kDm, kDm);
  k_attn<<<kBC * kNh, 64, 0, stream>>>(Qb, Kb, Vb, nullptr, ctxA);
  gemm(stream, ctxA, F(ALL_OW), F(ALL_OB), Lo, kRows, kDm, kDm, kDm, kDm, kDm);
  float* tpY = T5;  // ctxA dead
  gemm(stream, Lo, F(TP_W), F(TP_B), tpY, kRows, kDm, kDm, kDm, kDm, kDm);

  // ---- aLH attention (on He, post-softmax key gating) ----
  float* ctxH = T6;  // Lo dead
  gemm(stream, He, F(ALH_QW), F(ALH_QB), Qb, kRows, kDm, kDm, kDm, kDm, kDm);
  gemm(stream, He, F(ALH_KW), F(ALH_KB), Kb, kRows, kDm, kDm, kDm, kDm, kDm);
  gemm(stream, He, F(ALH_VW), F(ALH_VB), Vb, kRows, kDm, kDm, kDm, kDm, kDm);
  k_attn<<<kBC * kNh, 64, 0, stream>>>(Qb, Kb, Vb, gateA, ctxH);
  float* Ho = T7;
  gemm(stream, ctxH, F(ALH_OW), F(ALH_OB), Ho, kRows, kDm, kDm, kDm, kDm, kDm);
  float* gpY = T6;  // ctxH dead
  gemm(stream, Ho, F(GP_W), F(GP_B), gpY, kRows, kDm, kDm, kDm, kDm, kDm);

  // ---- fusion + pre-SSM LayerNorms ----
  float* fused = T1;  // Qb dead
  k_fused<<<kRows, 256, 0, stream>>>(tpY, gpY, LL, fused);
  float* xin1 = T0;  // LL dead
  k_ln<<<kRows, kDm, 0, stream>>>(fused, nullptr, F(N1_W), F(N1_B), nullptr, xin1);
  float* xin2 = T3;  // Kb dead
  k_ln<<<kRows, kDm, 0, stream>>>(Ho, He, F(N2_W), F(N2_B), nullptr, xin2);

  // ---- SSM 1 (on xin1) -> Lf ----
  float* xz = BIG0;   // emb dead
  float* proj = P0;
  float* delta = T2;  // He dead
  float* ssmT = T4;   // Vb dead
  float* ssmO = T5;   // tpY dead
  gemm(stream, xin1, F(S1_INW), nullptr, xz, kRows, 512, kDm, kDm, kDm, 512);
  gemm(stream, xz, F(S1_XW), nullptr, proj, kRows, 288, kDm, 512, kDm, 288);
  gemm(stream, proj, F(S1_DTW), F(S1_DTB), delta, kRows, kDm, kDm, 288, kDm, kDm);
  k_softplus<<<kRows, 256, 0, stream>>>(delta);
  k_scan<<<kBC, kDm, 0, stream>>>(delta, xz, proj, F(S1_ALOG), F(S1_D), ssmT);
  gemm(stream, ssmT, F(S1_OUTW), nullptr, ssmO, kRows, kDm, kDm, kDm, kDm, kDm);
  float* Lf = T6;  // gpY dead
  k_ln<<<kRows, kDm, 0, stream>>>(ssmO, xin1, F(S1_LNW), F(S1_LNB), fused, Lf);

  // ---- SSM 2 (on xin2) -> Hf ----
  gemm(stream, xin2, F(S2_INW), nullptr, xz, kRows, 512, kDm, kDm, kDm, 512);
  gemm(stream, xz, F(S2_XW), nullptr, proj, kRows, 288, kDm, 512, kDm, 288);
  gemm(stream, proj, F(S2_DTW), F(S2_DTB), delta, kRows, kDm, kDm, 288, kDm, kDm);
  k_softplus<<<kRows, 256, 0, stream>>>(delta);
  k_scan<<<kBC, kDm, 0, stream>>>(delta, xz, proj, F(S2_ALOG), F(S2_D), ssmT);
  gemm(stream, ssmT, F(S2_OUTW), nullptr, ssmO, kRows, kDm, kDm, kDm, kDm, kDm);
  float* Hf = T0;  // xin1 dead
  k_ln<<<kRows, kDm, 0, stream>>>(ssmO, xin2, F(S2_LNW), F(S2_LNB), Ho, Hf);

  // ---- iDWT -> head + FITS -> output ----
  float* recT = BIG0;  // xz dead
  k_idwt<<<kBC * kP, kDm, 0, stream>>>(Lf, Hf, F(DWT_H), F(DWT_G), recT);
  gemm(stream, recT, F(HEAD_W), F(HEAD_B), yhat, kBC, kPred, kDm * kP,
       kDm * kP, kDm * kP, kPred);
  k_dft<<<kBC, kCut, 0, stream>>>(xm, feat);
  gemm(stream, feat, F(FITS_W), F(FITS_B), fitsY, kBC, kPred, 2 * kCut,
       2 * kCut, 2 * kCut, kPred);
  k_final<<<kBC, kPred, 0, stream>>>(yhat, fitsY, F(RV_W), F(RV_B), meanA, stdA,
                                     (float*)d_out);
}
